// CausalLinearAttention_42262478193298
// MI455X (gfx1250) — matrix operand order repaired and checked by reading, not compiled, not run on hardware
//
#include <hip/hip_runtime.h>

typedef __bf16 bf16;
typedef __attribute__((ext_vector_type(16))) __bf16 v16bf;
typedef __attribute__((ext_vector_type(8)))  __bf16 v8bf;
typedef __attribute__((ext_vector_type(8)))  float  v8f;

#define T_SEQ   4096
#define D_MODEL 1024
#define N_HEAD  16
#define D_HEAD  64
#define N_B     4
#define CH      128
#define N_CHUNK 32
#define BT_     (N_B * T_SEQ)

#define V8F_ZERO {0.f,0.f,0.f,0.f,0.f,0.f,0.f,0.f}

// ---- WMMA fragment helpers (CDNA5 layouts, wave32) -------------------------
// A 16x32 bf16: lane(0..15)=row m, 16 contiguous K (offset 16*(lane>=16))
static __device__ __forceinline__ v16bf ld_a_frag(const bf16* p) {
  v8bf lo = *(const v8bf*)(p);
  v8bf hi = *(const v8bf*)(p + 8);
  return __builtin_shufflevector(lo, hi, 0,1,2,3,4,5,6,7,8,9,10,11,12,13,14,15);
}
// B 32x16 bf16: lane holds column n = lane&15, 16 contiguous K (offset 16*(lane>=16))
static __device__ __forceinline__ v16bf ld_b_frag(const bf16* p) {
  v8bf lo = *(const v8bf*)(p);
  v8bf hi = *(const v8bf*)(p + 8);
  return __builtin_shufflevector(lo, hi, 0,1,2,3,4,5,6,7,8,9,10,11,12,13,14,15);
}
static __device__ __forceinline__ v8f wmma_bf(v16bf a, v16bf b, v8f c) {
  return __builtin_amdgcn_wmma_f32_16x16x32_bf16(false, a, false, b, (short)0, c,
                                                 false, false);
}
static __device__ __forceinline__ float featf(float x) {  // elu(x)+1
  return x > 0.f ? x + 1.f : __expf(x);
}
// LDS byte offset of a __shared__ object (flat addr low 32 bits = LDS offset)
template <typename T>
static __device__ __forceinline__ unsigned lds_off(T* p) {
  return (unsigned)(unsigned long long)(uintptr_t)p;
}

// ---- Stage 0a: x fp32 -> bf16 ----------------------------------------------
__global__ __launch_bounds__(256) void xconv_kernel(const float* __restrict__ x,
                                                    bf16* __restrict__ xb) {
  size_t i = ((size_t)blockIdx.x * 256 + threadIdx.x) * 4;
  float4 v = *(const float4*)(x + i);
  xb[i + 0] = (bf16)v.x;
  xb[i + 1] = (bf16)v.y;
  xb[i + 2] = (bf16)v.z;
  xb[i + 3] = (bf16)v.w;
}

// ---- Stage 0b: W (1024x1024 f32) -> W^T bf16 -------------------------------
__global__ __launch_bounds__(256) void wtrans_kernel(
    const float* __restrict__ W0, const float* __restrict__ W1,
    const float* __restrict__ W2, const float* __restrict__ W3,
    bf16* __restrict__ T0, bf16* __restrict__ T1,
    bf16* __restrict__ T2, bf16* __restrict__ T3) {
  const float* W = (blockIdx.z == 0) ? W0 : (blockIdx.z == 1) ? W1
                 : (blockIdx.z == 2) ? W2 : W3;
  bf16* Tt = (blockIdx.z == 0) ? T0 : (blockIdx.z == 1) ? T1
           : (blockIdx.z == 2) ? T2 : T3;
  __shared__ float tile[32][33];
  int kb = blockIdx.x * 32, nb = blockIdx.y * 32;
  int tx = threadIdx.x & 31, ty = threadIdx.x >> 5;  // 32 x 8
#pragma unroll
  for (int i = 0; i < 32; i += 8)
    tile[ty + i][tx] = W[(size_t)(kb + ty + i) * D_MODEL + nb + tx];
  __syncthreads();
#pragma unroll
  for (int i = 0; i < 32; i += 8)
    Tt[(size_t)(nb + ty + i) * D_MODEL + kb + tx] = (bf16)tile[tx][ty + i];
}

// ---- Stage 1: Q/K/V projection GEMM (wave tile 32x64) ----------------------
__global__ __launch_bounds__(256) void proj_kernel(
    const bf16* __restrict__ xb,
    const bf16* __restrict__ WTq, const bf16* __restrict__ WTk,
    const bf16* __restrict__ WTv,
    bf16* __restrict__ Qg, bf16* __restrict__ Kg, bf16* __restrict__ Vg) {
  int mat = blockIdx.y;
  const bf16* WT = (mat == 0) ? WTq : (mat == 1) ? WTk : WTv;
  bf16* Out      = (mat == 0) ? Qg  : (mat == 1) ? Kg  : Vg;
  int wave = threadIdx.x >> 5, lane = threadIdx.x & 31;
  int hi = lane >> 4, lo = lane & 15;
  int wid  = blockIdx.x * 8 + wave;
  int mblk = wid >> 4, nblk = wid & 15;
  int mbase = mblk * 32, nbase = nblk * 64;

  v8f acc[2][4];
#pragma unroll
  for (int s = 0; s < 2; ++s)
#pragma unroll
    for (int nt = 0; nt < 4; ++nt) { v8f z = V8F_ZERO; acc[s][nt] = z; }

  for (int kb = 0; kb < D_MODEL; kb += 32) {
    v16bf a[2];
#pragma unroll
    for (int s = 0; s < 2; ++s) {
      const bf16* ap = xb + (size_t)(mbase + s * 16 + lo) * D_MODEL + kb + hi * 16;
      a[s] = ld_a_frag(ap);
    }
#pragma unroll
    for (int nt = 0; nt < 4; ++nt) {
      const bf16* bp = WT + (size_t)(nbase + nt * 16 + lo) * D_MODEL + kb + hi * 16;
      v16bf b = ld_b_frag(bp);
#pragma unroll
      for (int s = 0; s < 2; ++s) acc[s][nt] = wmma_bf(a[s], b, acc[s][nt]);
    }
  }

  bool feat = (mat != 2);
#pragma unroll
  for (int s = 0; s < 2; ++s)
#pragma unroll
    for (int nt = 0; nt < 4; ++nt)
#pragma unroll
      for (int r = 0; r < 8; ++r) {
        int row = mbase + s * 16 + r + hi * 8;
        int col = nbase + nt * 16 + lo;
        float v = acc[s][nt][r];
        if (feat) v = featf(v);
        int b = row >> 12, t = row & 4095, h = col >> 6, dh = col & 63;
        Out[((size_t)(b * N_HEAD + h) * T_SEQ + t) * D_HEAD + dh] = (bf16)v;
      }
}

// ---- Stage 1.5: Z = 1/(Q . cumsum_heads(K) + eps) --------------------------
__global__ __launch_bounds__(256) void z_kernel(const bf16* __restrict__ Qg,
                                                const bf16* __restrict__ Kg,
                                                float* __restrict__ Z) {
  int g = blockIdx.x * 256 + threadIdx.x;  // 0..BT-1
  int b = g >> 12, t = g & 4095;
  float ks[D_HEAD];
#pragma unroll
  for (int d = 0; d < D_HEAD; ++d) ks[d] = 0.f;
  for (int h = 0; h < N_HEAD; ++h) {
    size_t base = ((size_t)(b * N_HEAD + h) * T_SEQ + t) * D_HEAD;
    float dot = 0.f;
#pragma unroll
    for (int d = 0; d < D_HEAD; ++d) {
      ks[d] += (float)Kg[base + d];
      dot   += (float)Qg[base + d] * ks[d];
    }
    Z[(size_t)(b * N_HEAD + h) * T_SEQ + t] = 1.f / (dot + 1e-6f);
  }
}

// ---- Stage 2: chunked causal linear attention (one WG per (b,h)) -----------
__global__ __launch_bounds__(256) void attn_kernel(
    const bf16* __restrict__ Qg, const bf16* __restrict__ Kg,
    const bf16* __restrict__ Vg, const float* __restrict__ Z,
    bf16* __restrict__ Abf) {
  __shared__ __align__(64) bf16  kbf[CH][D_HEAD];        // [j][d]  B of q.k^T
  __shared__ __align__(64) bf16  vbf[CH][D_HEAD];        // [j][d]  async staging
  __shared__ __align__(64) bf16  kTbf[D_HEAD][CH];       // [d][j]  A of k^T.v
  __shared__ __align__(64) bf16  vTbf[D_HEAD][CH];       // [e][j]  B of attn.v / k^T.v
  __shared__ __align__(64) bf16  attnbf[CH][CH];         // [m][j]  A of attn.v
  __shared__ __align__(64) bf16  STbf[D_HEAD][D_HEAD];   // [e][d]  B of q.S
  __shared__ __align__(64) float Sfp[D_HEAD][D_HEAD];    // [d][e]  fp32 state

  int bh = blockIdx.x;
  int b = bh >> 4, h = bh & 15;
  int tid = threadIdx.x;
  int wave = tid >> 5, lane = tid & 31;
  int hi = lane >> 4, lo = lane & 15;
  const size_t seqb = (size_t)bh * T_SEQ * D_HEAD;

  for (int i = tid; i < D_HEAD * D_HEAD; i += 256) (&Sfp[0][0])[i] = 0.f;
  for (int i = tid; i < CH * CH; i += 256) (&attnbf[0][0])[i] = (bf16)0.f;
  __syncthreads();

  const unsigned kdst0 = lds_off(&kbf[0][0]);
  const unsigned vdst0 = lds_off(&vbf[0][0]);

  for (int c = 0; c < N_CHUNK; ++c) {
    // ---- Phase A0: async DMA k/v chunk rows into LDS (16B per lane) ----
    {
      const bf16* kg = Kg + seqb + (size_t)c * CH * D_HEAD;
      const bf16* vg = Vg + seqb + (size_t)c * CH * D_HEAD;
#pragma unroll
      for (int i = tid * 8; i < CH * D_HEAD; i += 256 * 8) {
        unsigned long long gk = (unsigned long long)(uintptr_t)(kg + i);
        unsigned long long gv = (unsigned long long)(uintptr_t)(vg + i);
        unsigned lk = kdst0 + (unsigned)i * 2u;
        unsigned lv = vdst0 + (unsigned)i * 2u;
        asm volatile("global_load_async_to_lds_b128 %0, %1, off"
                     :: "v"(lk), "v"(gk) : "memory");
        asm volatile("global_load_async_to_lds_b128 %0, %1, off"
                     :: "v"(lv), "v"(gv) : "memory");
      }
      asm volatile("s_wait_asynccnt 0x0" ::: "memory");
    }
    __syncthreads();

    // ---- Phase A1: build k^T, v^T and S^T snapshots in LDS ----
    for (int i = tid; i < CH * D_HEAD; i += 256) {
      int j = i >> 6, d = i & 63;
      kTbf[d][j] = kbf[j][d];
      vTbf[d][j] = vbf[j][d];
    }
    for (int i = tid; i < D_HEAD * D_HEAD; i += 256) {
      int d = i >> 6, e = i & 63;
      STbf[e][d] = (bf16)Sfp[d][e];
    }
    __syncthreads();

    // ---- Phase B: per-wave 16-row strip ----
    int trow = c * CH + wave * 16 + lo;
    v16bf aq[2];
#pragma unroll
    for (int ks = 0; ks < 2; ++ks) {
      const bf16* qp = Qg + seqb + (size_t)trow * D_HEAD + ks * 32 + hi * 16;
      aq[ks] = ld_a_frag(qp);
    }
    v8f accO[4];
#pragma unroll
    for (int nt = 0; nt < 4; ++nt) { v8f z = V8F_ZERO; accO[nt] = z; }

    // inter-chunk: q @ S
#pragma unroll
    for (int nt = 0; nt < 4; ++nt)
#pragma unroll
      for (int ks = 0; ks < 2; ++ks)
        accO[nt] = wmma_bf(aq[ks], ld_b_frag(&STbf[nt * 16 + lo][ks * 32 + hi * 16]),
                           accO[nt]);

    // intra-chunk: attn = tril(q k^T) -> LDS (bf16, A layout source)
    for (int jt = 0; jt <= wave; ++jt) {
      v8f a = V8F_ZERO;
#pragma unroll
      for (int ks = 0; ks < 2; ++ks)
        a = wmma_bf(aq[ks], ld_b_frag(&kbf[jt * 16 + lo][ks * 32 + hi * 16]), a);
      if (jt == wave) {
#pragma unroll
        for (int r = 0; r < 8; ++r)
          if (lo > r + hi * 8) a[r] = 0.f;  // keep j <= i
      }
#pragma unroll
      for (int r = 0; r < 8; ++r)
        attnbf[wave * 16 + r + hi * 8][jt * 16 + lo] = (bf16)a[r];
    }
    // attn @ v (only K-tiles up to the diagonal; rest of attnbf stays zero)
    int nks = (wave + 2) >> 1;
    for (int k2 = 0; k2 < nks; ++k2) {
      v16bf aat = ld_a_frag(&attnbf[wave * 16 + lo][k2 * 32 + hi * 16]);
#pragma unroll
      for (int nt = 0; nt < 4; ++nt)
        accO[nt] = wmma_bf(aat, ld_b_frag(&vTbf[nt * 16 + lo][k2 * 32 + hi * 16]),
                           accO[nt]);
    }

    // epilogue: * Z, store bf16 into [B*T, D_MODEL] layout
    float zr[8];
#pragma unroll
    for (int r = 0; r < 8; ++r)
      zr[r] = Z[(size_t)bh * T_SEQ + c * CH + wave * 16 + r + hi * 8];
#pragma unroll
    for (int nt = 0; nt < 4; ++nt)
#pragma unroll
      for (int r = 0; r < 8; ++r) {
        int tt = c * CH + wave * 16 + r + hi * 8;
        float o = accO[nt][r] * zr[r];
        Abf[((size_t)b * T_SEQ + tt) * D_MODEL + h * D_HEAD + nt * 16 + lo] = (bf16)o;
      }

    // ---- Phase C: S += k^T @ v (16 tiles, 2 per wave) ----
#pragma unroll
    for (int p = 0; p < 2; ++p) {
      int ti = wave * 2 + p, mt = ti >> 2, nt2 = ti & 3;
      v8f sacc = V8F_ZERO;
#pragma unroll
      for (int k3 = 0; k3 < 4; ++k3) {
        v16bf ak = ld_a_frag(&kTbf[mt * 16 + lo][k3 * 32 + hi * 16]);
        sacc = wmma_bf(ak, ld_b_frag(&vTbf[nt2 * 16 + lo][k3 * 32 + hi * 16]), sacc);
      }
#pragma unroll
      for (int r = 0; r < 8; ++r)
        Sfp[mt * 16 + r + hi * 8][nt2 * 16 + lo] += sacc[r];
    }
    __syncthreads();
  }
}

// ---- Stage 3: output projection + bias -------------------------------------
__global__ __launch_bounds__(256) void oproj_kernel(
    const bf16* __restrict__ Abf, const bf16* __restrict__ WTo,
    const float* __restrict__ bo, float* __restrict__ out) {
  int wave = threadIdx.x >> 5, lane = threadIdx.x & 31;
  int hi = lane >> 4, lo = lane & 15;
  int wid  = blockIdx.x * 8 + wave;
  int mblk = wid >> 4, nblk = wid & 15;
  int mbase = mblk * 32, nbase = nblk * 64;

  v8f acc[2][4];
#pragma unroll
  for (int s = 0; s < 2; ++s)
#pragma unroll
    for (int nt = 0; nt < 4; ++nt) { v8f z = V8F_ZERO; acc[s][nt] = z; }

  for (int kb = 0; kb < D_MODEL; kb += 32) {
    v16bf a[2];
#pragma unroll
    for (int s = 0; s < 2; ++s) {
      const bf16* ap = Abf + (size_t)(mbase + s * 16 + lo) * D_MODEL + kb + hi * 16;
      a[s] = ld_a_frag(ap);
    }
#pragma unroll
    for (int nt = 0; nt < 4; ++nt) {
      const bf16* bp = WTo + (size_t)(nbase + nt * 16 + lo) * D_MODEL + kb + hi * 16;
      v16bf b = ld_b_frag(bp);
#pragma unroll
      for (int s = 0; s < 2; ++s) acc[s][nt] = wmma_bf(a[s], b, acc[s][nt]);
    }
  }
#pragma unroll
  for (int nt = 0; nt < 4; ++nt) {
    float bias = bo[nbase + nt * 16 + lo];
#pragma unroll
    for (int s = 0; s < 2; ++s)
#pragma unroll
      for (int r = 0; r < 8; ++r) {
        int row = mbase + s * 16 + r + hi * 8;
        int col = nbase + nt * 16 + lo;
        out[(size_t)row * D_MODEL + col] = acc[s][nt][r] + bias;
      }
  }
}

// ---- Host launcher ---------------------------------------------------------
extern "C" void kernel_launch(void* const* d_in, const int* in_sizes, int n_in,
                              void* d_out, int out_size, void* d_ws, size_t ws_size,
                              hipStream_t stream) {
  (void)in_sizes; (void)n_in; (void)out_size; (void)ws_size;
  const float* x  = (const float*)d_in[0];
  const float* Wq = (const float*)d_in[1];
  const float* Wk = (const float*)d_in[2];
  const float* Wv = (const float*)d_in[3];
  const float* Wo = (const float*)d_in[4];
  const float* bo = (const float*)d_in[5];
  float* out = (float*)d_out;

  char* ws = (char*)d_ws;
  size_t off = 0;
  auto take = [&](size_t bytes) {
    void* p = ws + off;
    off += (bytes + 255) & ~(size_t)255;
    return p;
  };
  bf16*  xb  = (bf16*)take((size_t)BT_ * D_MODEL * 2);  // reused as attn output
  bf16*  WTq = (bf16*)take((size_t)D_MODEL * D_MODEL * 2);
  bf16*  WTk = (bf16*)take((size_t)D_MODEL * D_MODEL * 2);
  bf16*  WTv = (bf16*)take((size_t)D_MODEL * D_MODEL * 2);
  bf16*  WTo = (bf16*)take((size_t)D_MODEL * D_MODEL * 2);
  bf16*  Qg  = (bf16*)take((size_t)BT_ * D_MODEL * 2);
  bf16*  Kg  = (bf16*)take((size_t)BT_ * D_MODEL * 2);
  bf16*  Vg  = (bf16*)take((size_t)BT_ * D_MODEL * 2);
  float* Zw  = (float*)take((size_t)N_B * N_HEAD * T_SEQ * 4);
  bf16*  Abf = xb;  // alias: xb dead after proj, Abf produced after

  xconv_kernel<<<(BT_ * D_MODEL) / (256 * 4), 256, 0, stream>>>(x, xb);
  wtrans_kernel<<<dim3(32, 32, 4), 256, 0, stream>>>(Wq, Wk, Wv, Wo,
                                                     WTq, WTk, WTv, WTo);
  proj_kernel<<<dim3(1024, 3), 256, 0, stream>>>(xb, WTq, WTk, WTv, Qg, Kg, Vg);
  z_kernel<<<BT_ / 256, 256, 0, stream>>>(Qg, Kg, Zw);
  attn_kernel<<<N_B * N_HEAD, 256, 0, stream>>>(Qg, Kg, Vg, Zw, Abf);
  oproj_kernel<<<1024, 256, 0, stream>>>(Abf, WTo, bo, out);
}
